// SparseAttention_3685081940023
// MI455X (gfx1250) — compile-verified, matrix operand order checked by reading, not run on hardware
//
#include <hip/hip_runtime.h>

typedef __attribute__((ext_vector_type(8)))  float  v8f;
typedef __attribute__((ext_vector_type(16))) __bf16 v16bf;
typedef __attribute__((ext_vector_type(8)))  __bf16 v8bf;

#define HH    8
#define SS    64
#define KNB   32
#define CPD   128
#define CDIM  512
#define QKVW  1536   // H * 3S
#define DCAT  1536   // H*CP + H*S

// ---------------- prep kernels ----------------
__global__ void f32_to_bf16_kernel(const float* __restrict__ in, __bf16* __restrict__ out, int n) {
    int i = blockIdx.x * blockDim.x + threadIdx.x;
    if (i < n) out[i] = (__bf16)in[i];
}

// in[rows][cols] (f32) -> out[cols][rows] (bf16)
__global__ void transpose_to_bf16_kernel(const float* __restrict__ in, __bf16* __restrict__ out,
                                         int rows, int cols) {
    int i = blockIdx.x * blockDim.x + threadIdx.x;
    if (i < rows * cols) {
        int r = i / cols, c = i % cols;
        out[(size_t)c * rows + r] = (__bf16)in[i];
    }
}

// ---------------- bf16 WMMA GEMM:  C[M,N] = A[M,K] * Bt[N,K]^T (+bias) ----------------
// block = 256 threads (8 waves), block tile 128(M) x 128(N), K step 32, double-buffered LDS.
// Tiles stream via GLOBAL_LOAD_ASYNC_TO_LDS_B128 (ASYNCcnt), overlapped with the 8-WMMA
// block; s_wait_asynccnt 0 is pinned after the WMMAs and before the barrier so each trip's
// async copies hide under matrix math, and the barrier publishes completed LDS tiles.
template <bool HAS_BIAS>
__global__ __launch_bounds__(256)
void gemm_bf16_wmma_kernel(const __bf16* __restrict__ A, const __bf16* __restrict__ Bt,
                           const float* __restrict__ bias, float* __restrict__ C,
                           int M, int N, int K) {
    // rows padded to 40 halves (80B = 5x16B keeps b128 alignment, breaks bank conflicts)
    __shared__ __align__(16) __bf16 As[2][128 * 40];
    __shared__ __align__(16) __bf16 Bs[2][128 * 40];

    const int t     = threadIdx.x;
    const int lane  = t & 31;
    const int wave  = t >> 5;
    const int lo    = lane & 15;
    const int hi    = lane >> 4;
    const int waveM = wave & 3;   // 0..3 -> 32-row strip
    const int waveN = wave >> 2;  // 0..1 -> 64-col strip
    const int mTile = blockIdx.x * 128;
    const int nTile = blockIdx.y * 128;

    v8f acc[2][4];
#pragma unroll
    for (int a = 0; a < 2; ++a)
#pragma unroll
        for (int b = 0; b < 4; ++b)
            acc[a][b] = (v8f){0.f, 0.f, 0.f, 0.f, 0.f, 0.f, 0.f, 0.f};

    // global -> LDS async tile copy (A: 128x32, B: 128x32 bf16; 512 x b128 each, 2 per thread).
    // No VGPR data round-trip, no per-load s_wait_loadcnt: tracked on ASYNCcnt only.
    auto load_tiles = [&](int k0, int buf) {
#pragma unroll
        for (int it = 0; it < 2; ++it) {
            int idx = t + it * 256;
            int r = idx >> 2, seg = idx & 3;
            const __bf16* ga = A + (size_t)(mTile + r) * K + k0 + seg * 8;
            unsigned la = (unsigned)(unsigned long long)&As[buf][r * 40 + seg * 8];
            asm volatile("global_load_async_to_lds_b128 %0, %1, off"
                         :: "v"(la), "v"(ga) : "memory");
            const __bf16* gb = Bt + (size_t)(nTile + r) * K + k0 + seg * 8;
            unsigned lb = (unsigned)(unsigned long long)&Bs[buf][r * 40 + seg * 8];
            asm volatile("global_load_async_to_lds_b128 %0, %1, off"
                         :: "v"(lb), "v"(gb) : "memory");
        }
    };

    load_tiles(0, 0);
    asm volatile("s_wait_asynccnt 0x0" ::: "memory");
    __syncthreads();

    int cur = 0;
    for (int k0 = 0; k0 < K; k0 += 32) {
        if (k0 + 32 < K) load_tiles(k0 + 32, cur ^ 1);   // async, overlaps with WMMA below

        // Fragments per ISA layout: elems 0..7 -> K = hi*8 + j ; elems 8..15 -> K = 16 + hi*8 + (j-8)
        v16bf aF[2], bF[4];
#pragma unroll
        for (int mi = 0; mi < 2; ++mi) {
            int row = waveM * 32 + mi * 16 + lo;
            v8bf c0 = *reinterpret_cast<const v8bf*>(&As[cur][row * 40 + hi * 8]);
            v8bf c1 = *reinterpret_cast<const v8bf*>(&As[cur][row * 40 + 16 + hi * 8]);
            aF[mi] = __builtin_shufflevector(c0, c1, 0,1,2,3,4,5,6,7,8,9,10,11,12,13,14,15);
        }
#pragma unroll
        for (int ni = 0; ni < 4; ++ni) {
            int row = waveN * 64 + ni * 16 + lo;
            v8bf c0 = *reinterpret_cast<const v8bf*>(&Bs[cur][row * 40 + hi * 8]);
            v8bf c1 = *reinterpret_cast<const v8bf*>(&Bs[cur][row * 40 + 16 + hi * 8]);
            bF[ni] = __builtin_shufflevector(c0, c1, 0,1,2,3,4,5,6,7,8,9,10,11,12,13,14,15);
        }
#pragma unroll
        for (int mi = 0; mi < 2; ++mi)
#pragma unroll
            for (int ni = 0; ni < 4; ++ni)
                acc[mi][ni] = __builtin_amdgcn_wmma_f32_16x16x32_bf16(
                    false, aF[mi], false, bF[ni], (short)0, acc[mi][ni], false, false);

        // Pin the accumulators so the scheduler cannot hoist the async wait above the
        // WMMA block (volatile asm statements keep mutual program order).
        asm volatile("" : "+v"(acc[0][0]), "+v"(acc[0][1]), "+v"(acc[0][2]), "+v"(acc[0][3]),
                          "+v"(acc[1][0]), "+v"(acc[1][1]), "+v"(acc[1][2]), "+v"(acc[1][3]));
        // publish the freshly streamed tiles before the barrier releases consumers
        asm volatile("s_wait_asynccnt 0x0" ::: "memory");
        __syncthreads();
        cur ^= 1;
    }

    // epilogue: VGPR r -> row hi*8 + r, col = lo (per C/D layout); bias hoisted per column
#pragma unroll
    for (int ni = 0; ni < 4; ++ni) {
        int col = nTile + waveN * 64 + ni * 16 + lo;
        float bcol = HAS_BIAS ? bias[col] : 0.f;
#pragma unroll
        for (int mi = 0; mi < 2; ++mi)
#pragma unroll
            for (int r = 0; r < 8; ++r) {
                int row = mTile + waveM * 32 + mi * 16 + hi * 8 + r;
                C[(size_t)row * N + col] = acc[mi][ni][r] + bcol;
            }
    }
}

// ---------------- LayerNorm over q and k heads, in place ----------------
// one wave per (n, h): q = qkv[n, h*192 + 0..63], k = qkv[n, h*192 + 64..127]
__global__ __launch_bounds__(256)
void ln_qk_kernel(float* __restrict__ qkv, const float* __restrict__ g_q, const float* __restrict__ b_q,
                  const float* __restrict__ g_k, const float* __restrict__ b_k) {
    int gid  = blockIdx.x * 8 + (threadIdx.x >> 5);
    int lane = threadIdx.x & 31;
    int n = gid >> 3, h = gid & 7;
    float* base = qkv + (size_t)n * QKVW + h * 192;
#pragma unroll
    for (int part = 0; part < 2; ++part) {
        float* p = base + part * 64;
        const float* g = part ? g_k : g_q;
        const float* b = part ? b_k : b_q;
        float x0 = p[lane], x1 = p[lane + 32];
        float s = x0 + x1, s2 = x0 * x0 + x1 * x1;
#pragma unroll
        for (int off = 16; off; off >>= 1) {
            s  += __shfl_xor(s,  off, 32);
            s2 += __shfl_xor(s2, off, 32);
        }
        float mean = s * (1.f / 64.f);
        float var  = s2 * (1.f / 64.f) - mean * mean;
        float inv  = rsqrtf(var + 1e-5f);
        p[lane]      = (x0 - mean) * inv * g[lane]      + b[lane];
        p[lane + 32] = (x1 - mean) * inv * g[lane + 32] + b[lane + 32];
    }
}

// ---------------- fused per-node attention: bias + dot + softmax + out_pair/out_scalar ----------------
__global__ __launch_bounds__(256)
void attn_kernel(const float* __restrict__ qkv, const float* __restrict__ pair,
                 const int* __restrict__ neighbours, const unsigned char* __restrict__ mask,
                 const float* __restrict__ w_bias, __bf16* __restrict__ cat) {
    __shared__ __align__(16) float sp[KNB][CPD];   // 16 KB pair tile
    __shared__ float sq[HH][SS];                   // 2 KB  q rows (post-LN)
    __shared__ float swb[CPD][HH];                 // 4 KB  w_bias
    __shared__ float sattn[KNB][HH];               // logits, then attn
    __shared__ int   snbr[KNB];
    __shared__ int   svalid[KNB];

    const int i = blockIdx.x;
    const int t = threadIdx.x;

    {   // pair tile (float4 copies)
        const float4* src = reinterpret_cast<const float4*>(pair + (size_t)i * KNB * CPD);
        float4* dst = reinterpret_cast<float4*>(&sp[0][0]);
        for (int j = t; j < KNB * CPD / 4; j += 256) dst[j] = src[j];
    }
    for (int j = t; j < HH * SS; j += 256) {
        int h = j >> 6, s = j & 63;
        sq[h][s] = qkv[(size_t)i * QKVW + h * 192 + s];
    }
    for (int j = t; j < CPD * HH; j += 256) (&swb[0][0])[j] = w_bias[j];
    if (t < KNB) {
        int nb = neighbours[(size_t)i * KNB + t];
        snbr[t]   = nb;
        svalid[t] = (mask[(size_t)i * KNB + t] != 0) && (nb != -1);
    }
    __syncthreads();

    {   // logits[k][h] : one (k,h) per thread
        int k = t >> 3, h = t & 7;
        float logit = -1e9f;
        if (svalid[k]) {
            float bias = 0.f;
            for (int c = 0; c < CPD; ++c) bias += sp[k][c] * swb[c][h];
            const float* krow = qkv + (size_t)snbr[k] * QKVW + h * 192 + 64;  // L2-resident gather
            float dot = 0.f;
            for (int s = 0; s < SS; ++s) dot += sq[h][s] * krow[s];
            logit = 0.70710678f * (dot * 0.125f + bias);   // 1/sqrt(64)=0.125, w_L=sqrt(0.5)
        }
        sattn[k][h] = logit;
    }
    __syncthreads();

    if (t < HH) {   // softmax over K per head
        int h = t;
        float mx = -1e30f;
        for (int k = 0; k < KNB; ++k) mx = fmaxf(mx, sattn[k][h]);
        float e[KNB], sum = 0.f;
        for (int k = 0; k < KNB; ++k) { e[k] = __expf(sattn[k][h] - mx); sum += e[k]; }
        float r = 1.f / sum;
        for (int k = 0; k < KNB; ++k) sattn[k][h] = svalid[k] ? e[k] * r : 0.f;
    }
    __syncthreads();

    {   // out_pair[h][c] -> cat[i, h*128 + c] (bf16)
        int h = t >> 5, c0 = (t & 31) * 4;
        float o0 = 0.f, o1 = 0.f, o2 = 0.f, o3 = 0.f;
        for (int k = 0; k < KNB; ++k) {
            float a = sattn[k][h];
            o0 += a * sp[k][c0 + 0]; o1 += a * sp[k][c0 + 1];
            o2 += a * sp[k][c0 + 2]; o3 += a * sp[k][c0 + 3];
        }
        __bf16* dst = cat + (size_t)i * DCAT + h * CPD + c0;
        dst[0] = (__bf16)o0; dst[1] = (__bf16)o1; dst[2] = (__bf16)o2; dst[3] = (__bf16)o3;
    }
    // out_scalar[h][s] -> cat[i, 1024 + h*64 + s] (bf16)
    for (int idx = t; idx < HH * SS; idx += 256) {
        int h = idx >> 6, s = idx & 63;
        float o = 0.f;
        for (int k = 0; k < KNB; ++k) {
            if (svalid[k]) {
                float a = sattn[k][h];
                o += a * qkv[(size_t)snbr[k] * QKVW + h * 192 + 128 + s];
            }
        }
        cat[(size_t)i * DCAT + HH * CPD + h * SS + s] = (__bf16)o;
    }
}

// ---------------- launch ----------------
extern "C" void kernel_launch(void* const* d_in, const int* in_sizes, int n_in,
                              void* d_out, int out_size, void* d_ws, size_t ws_size,
                              hipStream_t stream) {
    const float*         local      = (const float*)d_in[0];
    const float*         pair       = (const float*)d_in[1];
    const int*           neighbours = (const int*)d_in[2];
    const unsigned char* mask       = (const unsigned char*)d_in[3];
    const float*         w_qkv      = (const float*)d_in[4];
    const float*         g_q        = (const float*)d_in[5];
    const float*         b_q        = (const float*)d_in[6];
    const float*         g_k        = (const float*)d_in[7];
    const float*         b_k        = (const float*)d_in[8];
    const float*         w_bias     = (const float*)d_in[9];
    const float*         w_out      = (const float*)d_in[10];
    const float*         b_out      = (const float*)d_in[11];
    float*               out        = (float*)d_out;

    const int N = in_sizes[0] / CDIM;   // 16384

    auto align256 = [](size_t x) { return (x + 255) & ~(size_t)255; };
    char* ws = (char*)d_ws;
    size_t off = 0;
    __bf16* local_bf = (__bf16*)(ws + off); off = align256(off + (size_t)N * CDIM * 2);
    __bf16* wqkvT    = (__bf16*)(ws + off); off = align256(off + (size_t)CDIM * QKVW * 2);
    __bf16* woutT    = (__bf16*)(ws + off); off = align256(off + (size_t)DCAT * CDIM * 2);
    float*  qkv      = (float*) (ws + off); off = align256(off + (size_t)N * QKVW * 4);
    __bf16* cat      = (__bf16*)(ws + off);

    {   int n = N * CDIM;
        f32_to_bf16_kernel<<<(n + 255) / 256, 256, 0, stream>>>(local, local_bf, n); }
    {   int n = CDIM * QKVW;
        transpose_to_bf16_kernel<<<(n + 255) / 256, 256, 0, stream>>>(w_qkv, wqkvT, CDIM, QKVW); }
    {   int n = DCAT * CDIM;
        transpose_to_bf16_kernel<<<(n + 255) / 256, 256, 0, stream>>>(w_out, woutT, DCAT, CDIM); }

    // qkv = local @ w_qkv   [N,512] x [512,1536]
    gemm_bf16_wmma_kernel<false><<<dim3(N / 128, QKVW / 128), 256, 0, stream>>>(
        local_bf, wqkvT, nullptr, qkv, N, QKVW, CDIM);

    // LayerNorm q,k in place
    ln_qk_kernel<<<N * HH / 8, 256, 0, stream>>>(qkv, g_q, b_q, g_k, b_k);

    // fused attention -> cat (bf16)
    attn_kernel<<<N, 256, 0, stream>>>(qkv, pair, neighbours, mask, w_bias, cat);

    // out = cat @ w_out + b_out   [N,1536] x [1536,512]
    gemm_bf16_wmma_kernel<true><<<dim3(N / 128, CDIM / 128), 256, 0, stream>>>(
        cat, woutT, b_out, out, N, CDIM, DCAT);
}
